// MobilityGNNLayer_53532472377744
// MI455X (gfx1250) — matrix operand order repaired and checked by reading, not compiled, not run on hardware
//
#include <hip/hip_runtime.h>
#include <hip/hip_bf16.h>

// ---------------------------------------------------------------------------
// MobilityGNNLayer for MI455X (gfx1250, wave32, WMMA bf16 16x16x32)
// N=8192 nodes, D=256 features.
// Dominant cost: num = Wm^T @ T  (34.4 GFLOP, streams 256MB of M once).
// LDS tiles are laid out so every WMMA fragment load is 2x ds_load_b128.
// ---------------------------------------------------------------------------

#define NN      8192
#define DD      256
#define EPS_    1e-8f
#define THR_    1e-6f
#define LN_EPS_ 1e-5f

typedef __attribute__((ext_vector_type(16))) __bf16 v16bf;
typedef __attribute__((ext_vector_type(8)))  __bf16 v8bf;
typedef __attribute__((ext_vector_type(8)))  float  v8f;

__device__ __forceinline__ unsigned short f2bfu(float f) {
    unsigned u = __builtin_bit_cast(unsigned, f);
    u += 0x7FFFu + ((u >> 16) & 1u);              // round-to-nearest-even
    return (unsigned short)(u >> 16);
}
__device__ __forceinline__ __bf16 f2bf(float f) {
    unsigned short h = f2bfu(f);
    return __builtin_bit_cast(__bf16, h);
}

__device__ __forceinline__ v8f zero8() {
    v8f z;
#pragma unroll
    for (int e = 0; e < 8; ++e) z[e] = 0.f;
    return z;
}

// Pitches in halves: rows are 80 bytes -> every fragment chunk 16B aligned.
#define KP  40   // A tile pitch: As[m*KP + k], k = 0..31
#define BPT 40   // B tile pitch (TRANSPOSED): Bt[n*BPT + k], k = 0..31

// A fragment, 16x32 bf16. Lane L<16: row M=L, K={0..7,16..23};
// lane L>=16: row M=L-16, K={8..15,24..31}. Two 16B LDS loads.
__device__ __forceinline__ v16bf load_a_frag(const __bf16* As, int mBase,
                                             int lane) {
    const __bf16* row = As + (mBase + (lane & 15)) * KP + ((lane >> 4) << 3);
    v8bf c0 = *(const v8bf*)(row);
    v8bf c1 = *(const v8bf*)(row + 16);
    v16bf a;
#pragma unroll
    for (int e = 0; e < 8; ++e) { a[e] = c0[e]; a[e + 8] = c1[e]; }
    return a;
}

// B fragment, 32x16 bf16 from TRANSPOSED tile Bt[n][k].
// Lane L<16: N=L, K={0..7,16..23}; lane L>=16: N=L-16, K={8..15,24..31}.
__device__ __forceinline__ v16bf load_b_frag(const __bf16* Bt, int nBase,
                                             int lane) {
    const __bf16* p = Bt + (nBase + (lane & 15)) * BPT + ((lane >> 4) << 3);
    v8bf c0 = *(const v8bf*)(p);
    v8bf c1 = *(const v8bf*)(p + 16);
    v16bf b;
#pragma unroll
    for (int e = 0; e < 8; ++e) { b[e] = c0[e]; b[e + 8] = c1[e]; }
    return b;
}

__device__ __forceinline__ v8f wmma_bf16(v16bf a, v16bf b, v8f c) {
    return __builtin_amdgcn_wmma_f32_16x16x32_bf16(
        false, a, false, b, (short)0, c, false, false);
}

// ---------------------------------------------------------------------------

__global__ void zero_f32_kernel(float* p, int n) {
    int i = blockIdx.x * blockDim.x + threadIdx.x;
    if (i < n) p[i] = 0.f;
}

// colsum[i] = sum_j M[j,i].  Grid (32 col-tiles, 32 row-tiles), coalesced.
__global__ __launch_bounds__(256) void colsum_kernel(
    const float* __restrict__ M, float* __restrict__ colsum) {
    int i = blockIdx.x * 256 + threadIdx.x;
    int r0 = blockIdx.y * 256;
    float s = 0.f;
    for (int r = 0; r < 256; ++r) s += M[(r0 + r) * NN + i];
    atomicAdd(&colsum[i], s);
}

// ---------------------------------------------------------------------------
// T = NF @ W_in + b_in.  BM=64, BN=256 (full D), K-step 32. 8 waves,
// wave w: m-tile (w&3), n-tiles (w>>2)*8 .. +7  (8 accumulators).
// Emits T in f32 (fallback rows) and bf16 (for the big GEMM).
// ---------------------------------------------------------------------------
__global__ __launch_bounds__(256) void gemm_in_kernel(
    const float* __restrict__ NF, const float* __restrict__ Win,
    const float* __restrict__ bin, float* __restrict__ T32,
    __bf16* __restrict__ Tb) {
    __shared__ __bf16 As[64 * KP];
    __shared__ __bf16 Bt[DD * BPT];
    int t = threadIdx.x, lane = t & 31, w = t >> 5;
    int i0 = blockIdx.x * 64;
    int mt = (w & 3) * 16;
    int ng = (w >> 2) * 8;

    v8f acc[8];
#pragma unroll
    for (int j = 0; j < 8; ++j) acc[j] = zero8();

    for (int k0 = 0; k0 < DD; k0 += 32) {
        // A tile: NF rows are contiguous in k -> direct layout.
        for (int e = t; e < 64 * 32; e += 256) {
            int il = e >> 5, kl = e & 31;
            As[il * KP + kl] = f2bf(NF[(i0 + il) * DD + k0 + kl]);
        }
        // B tile, transposed: thread packs 4 consecutive k of one column d.
        for (int e = t; e < 8 * DD; e += 256) {
            int d = e & 255, kq = e >> 8;
            int kb = k0 + kq * 4;
            unsigned long long pk =
                (unsigned long long)f2bfu(Win[(kb + 0) * DD + d]) |
                ((unsigned long long)f2bfu(Win[(kb + 1) * DD + d]) << 16) |
                ((unsigned long long)f2bfu(Win[(kb + 2) * DD + d]) << 32) |
                ((unsigned long long)f2bfu(Win[(kb + 3) * DD + d]) << 48);
            *(unsigned long long*)(Bt + d * BPT + kq * 4) = pk;
        }
        __syncthreads();
        v16bf a = load_a_frag(As, mt, lane);
#pragma unroll
        for (int j = 0; j < 8; ++j) {
            v16bf b = load_b_frag(Bt, (ng + j) * 16, lane);
            acc[j] = wmma_bf16(a, b, acc[j]);
        }
        __syncthreads();
    }
    int hi8 = (lane >> 4) * 8, n0 = lane & 15;
#pragma unroll
    for (int j = 0; j < 8; ++j) {
        int n = (ng + j) * 16 + n0;
        float bb = bin[n];
#pragma unroll
        for (int r = 0; r < 8; ++r) {
            int i = i0 + mt + r + hi8;
            float v = acc[j][r] + bb;
            T32[i * DD + n] = v;
            Tb[i * DD + n] = f2bf(v);
        }
    }
}

// ---------------------------------------------------------------------------
// num = Wm^T @ T (the 34 GFLOP GEMM), fused with normalization, threshold,
// bf16 conversion and exact fp32 wsum accumulation. Each block owns 32
// destination nodes x all 256 features, so M streams from HBM exactly once;
// T-bf16 (4 MB) is L2-resident (192 MB L2).
// ---------------------------------------------------------------------------
__global__ __launch_bounds__(256) void gemm_big_kernel(
    const float* __restrict__ M, const __bf16* __restrict__ Tb,
    const float* __restrict__ colsum, float* __restrict__ num,
    float* __restrict__ wsum) {
    __shared__ __bf16 As[32 * KP];
    __shared__ __bf16 Bt[DD * BPT];
    __shared__ float csr[32];
    __shared__ float ws_s[256];
    int t = threadIdx.x, lane = t & 31, w = t >> 5;
    int i0 = blockIdx.x * 32;
    if (t < 32) csr[t] = 1.f / (colsum[i0 + t] + EPS_);
    __syncthreads();

    int mt = (w & 1) * 16;
    int nb = (w >> 1) * 4;
    v8f acc[4];
#pragma unroll
    for (int j = 0; j < 4; ++j) acc[j] = zero8();
    float wacc = 0.f; // this thread always covers column il = t&31

    const unsigned* Tb32 = (const unsigned*)Tb;

    for (int j0 = 0; j0 < NN; j0 += 32) {
        // A tile: coalesced M read (consecutive lanes -> consecutive i),
        // normalize + threshold + bf16, transposed scalar stores.
#pragma unroll
        for (int e = t; e < 32 * 32; e += 256) {
            int il = e & 31, jl = e >> 5;
            float wv = M[(j0 + jl) * NN + i0 + il] * csr[il];
            wv = (wv > THR_) ? wv : 0.f;
            wacc += wv;
            As[il * KP + jl] = f2bf(wv);
            if (j0 + 32 < NN)
                __builtin_prefetch(&M[(j0 + 32 + jl) * NN + i0 + il], 0, 0);
        }
        // B tile, transposed: 4 dword reads (4 k-rows, 2 d each) -> 2 b64.
#pragma unroll
        for (int e = t; e < 1024; e += 256) {
            int du = e & 127, kq = e >> 7;
            int kb = j0 + kq * 4;
            unsigned r0 = Tb32[(kb + 0) * 128 + du];
            unsigned r1 = Tb32[(kb + 1) * 128 + du];
            unsigned r2 = Tb32[(kb + 2) * 128 + du];
            unsigned r3 = Tb32[(kb + 3) * 128 + du];
            unsigned long long lo =
                (unsigned long long)((r0 & 0xffffu) | (r1 << 16)) |
                ((unsigned long long)((r2 & 0xffffu) | (r3 << 16)) << 32);
            unsigned long long hi =
                (unsigned long long)((r0 >> 16) | (r1 & 0xffff0000u)) |
                ((unsigned long long)((r2 >> 16) | (r3 & 0xffff0000u)) << 32);
            *(unsigned long long*)(Bt + (2 * du + 0) * BPT + kq * 4) = lo;
            *(unsigned long long*)(Bt + (2 * du + 1) * BPT + kq * 4) = hi;
        }
        __syncthreads();
        v16bf a = load_a_frag(As, mt, lane);
#pragma unroll
        for (int j = 0; j < 4; ++j) {
            v16bf b = load_b_frag(Bt, (nb + j) * 16, lane);
            acc[j] = wmma_bf16(a, b, acc[j]);
        }
        __syncthreads();
    }

    int hi8 = (lane >> 4) * 8, n0 = lane & 15;
#pragma unroll
    for (int j = 0; j < 4; ++j) {
        int n = (nb + j) * 16 + n0;
#pragma unroll
        for (int r = 0; r < 8; ++r)
            num[(i0 + mt + r + hi8) * DD + n] = acc[j][r];
    }
    ws_s[t] = wacc;
    __syncthreads();
    if (t < 32) {
        float s = 0.f;
#pragma unroll
        for (int q = 0; q < 8; ++q) s += ws_s[t + 32 * q];
        wsum[i0 + t] = s; // wsum > 0  <=>  has_edge (entries are 0 or > THR)
    }
}

// ---------------------------------------------------------------------------
// out_pre = agg @ W_out + b_out + NF, with agg row = has_edge ? num/(wsum+eps)
// : T, folded into A-staging.
// ---------------------------------------------------------------------------
__global__ __launch_bounds__(256) void gemm_out_kernel(
    const float* __restrict__ num, const float* __restrict__ T32,
    const float* __restrict__ wsum, const float* __restrict__ Wout,
    const float* __restrict__ bout, const float* __restrict__ NF,
    float* __restrict__ outp) {
    __shared__ __bf16 As[64 * KP];
    __shared__ __bf16 Bt[DD * BPT];
    __shared__ float rs[64]; // reciprocal wsum, or -1 => use T row
    int t = threadIdx.x, lane = t & 31, w = t >> 5;
    int i0 = blockIdx.x * 64;
    if (t < 64) {
        float s = wsum[i0 + t];
        rs[t] = (s > 0.f) ? 1.f / (s + EPS_) : -1.f;
    }
    __syncthreads();

    int mt = (w & 3) * 16;
    int ng = (w >> 2) * 8;
    v8f acc[8];
#pragma unroll
    for (int j = 0; j < 8; ++j) acc[j] = zero8();

    for (int k0 = 0; k0 < DD; k0 += 32) {
        for (int e = t; e < 64 * 32; e += 256) {
            int il = e >> 5, kl = e & 31;
            float sc = rs[il];
            int idx = (i0 + il) * DD + k0 + kl;
            float v = (sc < 0.f) ? T32[idx] : num[idx] * sc;
            As[il * KP + kl] = f2bf(v);
        }
        for (int e = t; e < 8 * DD; e += 256) {
            int d = e & 255, kq = e >> 8;
            int kb = k0 + kq * 4;
            unsigned long long pk =
                (unsigned long long)f2bfu(Wout[(kb + 0) * DD + d]) |
                ((unsigned long long)f2bfu(Wout[(kb + 1) * DD + d]) << 16) |
                ((unsigned long long)f2bfu(Wout[(kb + 2) * DD + d]) << 32) |
                ((unsigned long long)f2bfu(Wout[(kb + 3) * DD + d]) << 48);
            *(unsigned long long*)(Bt + d * BPT + kq * 4) = pk;
        }
        __syncthreads();
        v16bf a = load_a_frag(As, mt, lane);
#pragma unroll
        for (int j = 0; j < 8; ++j) {
            v16bf b = load_b_frag(Bt, (ng + j) * 16, lane);
            acc[j] = wmma_bf16(a, b, acc[j]);
        }
        __syncthreads();
    }
    int hi8 = (lane >> 4) * 8, n0 = lane & 15;
#pragma unroll
    for (int j = 0; j < 8; ++j) {
        int n = (ng + j) * 16 + n0;
        float bb = bout[n];
#pragma unroll
        for (int r = 0; r < 8; ++r) {
            int i = i0 + mt + r + hi8;
            outp[i * DD + n] = acc[j][r] + bb + NF[i * DD + n];
        }
    }
}

// ---------------------------------------------------------------------------
// LayerNorm over D=256, one wave32 per row (8 elements/lane + shfl_xor tree).
// ---------------------------------------------------------------------------
__global__ __launch_bounds__(256) void ln_kernel(
    const float* __restrict__ X, const float* __restrict__ g,
    const float* __restrict__ b, float* __restrict__ out) {
    int lane = threadIdx.x & 31, w = threadIdx.x >> 5;
    int row = blockIdx.x * 8 + w;
    const float* x = X + row * DD;
    float v[8], s = 0.f, s2 = 0.f;
#pragma unroll
    for (int q = 0; q < 8; ++q) {
        v[q] = x[lane + 32 * q];
        s += v[q];
        s2 += v[q] * v[q];
    }
#pragma unroll
    for (int off = 16; off >= 1; off >>= 1) {
        s += __shfl_xor(s, off, 32);
        s2 += __shfl_xor(s2, off, 32);
    }
    float mu = s * (1.f / DD);
    float var = s2 * (1.f / DD) - mu * mu;
    float rstd = rsqrtf(var + LN_EPS_);
#pragma unroll
    for (int q = 0; q < 8; ++q) {
        int e = lane + 32 * q;
        out[row * DD + e] = (v[q] - mu) * rstd * g[e] + b[e];
    }
}

// ---------------------------------------------------------------------------

extern "C" void kernel_launch(void* const* d_in, const int* in_sizes, int n_in,
                              void* d_out, int out_size, void* d_ws,
                              size_t ws_size, hipStream_t stream) {
    const float* NF   = (const float*)d_in[0]; // node_features [N,D]
    const float* M    = (const float*)d_in[1]; // mobility_matrix [N,N]
    const float* Win  = (const float*)d_in[2];
    const float* bin  = (const float*)d_in[3];
    const float* Wout = (const float*)d_in[4];
    const float* bout = (const float*)d_in[5];
    const float* lng  = (const float*)d_in[6];
    const float* lnb  = (const float*)d_in[7];
    float* out = (float*)d_out;

    // Workspace layout (~28 MB total)
    float*  colsum = (float*)d_ws;                              // NN f32
    float*  wsum   = colsum + NN;                               // NN f32
    float*  T32    = wsum + NN;                                 // NN*DD f32
    __bf16* Tb     = (__bf16*)(T32 + NN * DD);                  // NN*DD bf16
    float*  num    = (float*)((char*)Tb + (size_t)NN * DD * 2); // NN*DD f32
    float*  outp   = num + NN * DD;                             // NN*DD f32

    zero_f32_kernel<<<NN / 256, 256, 0, stream>>>(colsum, NN);
    colsum_kernel<<<dim3(NN / 256, NN / 256), 256, 0, stream>>>(M, colsum);
    gemm_in_kernel<<<NN / 64, 256, 0, stream>>>(NF, Win, bin, T32, Tb);
    gemm_big_kernel<<<NN / 32, 256, 0, stream>>>(M, Tb, colsum, num, wsum);
    gemm_out_kernel<<<NN / 64, 256, 0, stream>>>(num, T32, wsum, Wout, bout, NF, outp);
    ln_kernel<<<NN / 8, 256, 0, stream>>>(outp, lng, lnb, out);
}
